// SumProdLayer_83262236000759
// MI455X (gfx1250) — matrix operand order repaired and checked by reading, not compiled, not run on hardware
//
#include <hip/hip_runtime.h>

// CDNA5 / gfx1250 — wave32, WMMA 16x16x32 f16->f32
typedef _Float16 v16h __attribute__((ext_vector_type(16)));
typedef _Float16 v8h  __attribute__((ext_vector_type(8)));
typedef float    v8f  __attribute__((ext_vector_type(8)));

#define N_  64   // batch
#define CI  32   // in channels  (= WMMA K)
#define CO  32   // out channels
#define H_  32
#define W_  32
#define HO  16
#define WO  16

// One workgroup = one 2x2 output pixel block (4 input pixels).
// 8 waves (256 threads): wave w owns output tile (mtile = w>>1, ntile = w&1),
// i.e. rows n = 16*mtile..+15, cols o = 16*ntile..+15, looping the 4 pixels.
__global__ __launch_bounds__(256)
void sumprod_wmma_kernel(const float* __restrict__ x,
                         const float* __restrict__ logits,
                         float* __restrict__ out) {
  // A[p][n][i] = exp(x[n,i,h_p,w_p]) as f16   (4 * 64*32 * 2B = 16 KB)
  // B[p][o][i] = softmax_i(logits[o,:,h_p,w_p]) as f16 (4 * 32*32 * 2B = 8 KB)
  __shared__ __align__(16) _Float16 As[4][N_][CI];
  __shared__ __align__(32) _Float16 Bs[4][CO][CI];

  const int tid = threadIdx.x;
  const int blk = blockIdx.x;          // 0..255
  const int hb  = blk >> 4;            // output row block
  const int wb  = blk & 15;            // output col block
  const int h0  = hb * 2, w0 = wb * 2;

  // ---- Phase 1a: per-(o,pixel) softmax over I (threads 0..127) ----
  if (tid < 128) {
    const int o = tid & 31;
    const int p = tid >> 5;
    const int h = h0 + (p >> 1), w = w0 + (p & 1);
    const float* lp = logits + (size_t)o * (CI * H_ * W_) + h * W_ + w;
    float v[CI];
    float m = -3.402823466e38f;
#pragma unroll
    for (int i = 0; i < CI; ++i) { v[i] = lp[i * (H_ * W_)]; m = fmaxf(m, v[i]); }
    float s = 0.0f;
#pragma unroll
    for (int i = 0; i < CI; ++i) { v[i] = __expf(v[i] - m); s += v[i]; }
    const float inv = 1.0f / s;
#pragma unroll
    for (int i = 0; i < CI; ++i) Bs[p][o][i] = (_Float16)(v[i] * inv);
  }

  // ---- Phase 1b: A = exp(x) for the 4 pixels (all 256 threads) ----
  for (int j = tid; j < 4 * N_ * CI; j += 256) {
    const int p = j & 3;
    const int i = (j >> 2) & 31;
    const int n = j >> 7;
    const int h = h0 + (p >> 1), w = w0 + (p & 1);
    const float xv = x[(((size_t)n * CI + i) * H_ + h) * W_ + w];
    As[p][n][i] = (_Float16)__expf(xv);
  }

  __syncthreads();

  // ---- Phase 2: per-pixel GEMM via WMMA, accumulate log() (pooling) ----
  const int lane  = tid & 31;
  const int wave  = tid >> 5;
  const int mtile = wave >> 1;   // 0..3 -> n block
  const int ntile = wave & 1;    // 0..1 -> o block
  const int lrow  = lane & 15;
  const int lhi   = lane >> 4;   // lane half selects K-group / M-group

  v8f acc = {};                  // pooled log-sums for this lane's 8 (n,o) slots
#pragma unroll
  for (int p = 0; p < 4; ++p) {
    // A fragment (contiguous K order, same as B): lanes 0-15 hold K{0..15},
    // lanes 16-31 hold K{16..31}, row M = lane&15.
    const _Float16* arow = &As[p][mtile * 16 + lrow][0];
    v8h a0 = *(const v8h*)(arow + lhi * 16);       // K = 16*lhi .. +7
    v8h a1 = *(const v8h*)(arow + lhi * 16 + 8);   // K = 16*lhi + 8 .. +7
    v16h a = __builtin_shufflevector(a0, a1,
        0, 1, 2, 3, 4, 5, 6, 7, 8, 9, 10, 11, 12, 13, 14, 15);

    // B fragment (32x16): lane holds column N = lane&15,
    // K = 16*(lane>>4) .. +15 contiguous since Bs is [o][i].
    v16h b = *(const v16h*)(&Bs[p][ntile * 16 + lrow][lhi * 16]);

    v8f c = {};
    c = __builtin_amdgcn_wmma_f32_16x16x32_f16(
        /*neg_a=*/false, a, /*neg_b=*/false, b,
        /*c_mod=*/(short)0, c, /*reuse_a=*/false, /*reuse_b=*/false);

#pragma unroll
    for (int r = 0; r < 8; ++r) acc[r] += __logf(c[r]);
  }

  // ---- Phase 3: store pooled result; each (n,o,hb,wb) written exactly once ----
#pragma unroll
  for (int r = 0; r < 8; ++r) {
    const int n = mtile * 16 + lhi * 8 + r;  // C/D layout: VGPR r -> M = r or 8+r
    const int o = ntile * 16 + lrow;         // N = lane&15
    out[(((size_t)n * CO + o) * HO + hb) * WO + wb] = acc[r];
  }
}

extern "C" void kernel_launch(void* const* d_in, const int* in_sizes, int n_in,
                              void* d_out, int out_size, void* d_ws, size_t ws_size,
                              hipStream_t stream) {
  const float* x      = (const float*)d_in[0];  // [64,32,32,32] f32
  const float* logits = (const float*)d_in[1];  // [32,32,32,32] f32
  float* out          = (float*)d_out;          // [64,32,16,16] f32
  (void)in_sizes; (void)n_in; (void)out_size; (void)d_ws; (void)ws_size;

  dim3 grid(HO * WO);   // 256 workgroups, one per 2x2 pixel block
  dim3 block(256);      // 8 wave32 waves
  hipLaunchKernelGGL(sumprod_wmma_kernel, grid, block, 0, stream, x, logits, out);
}
